// NeuralMem_17849884082931
// MI455X (gfx1250) — matrix operand order repaired and checked by reading, not compiled, not run on hardware
//
#include <hip/hip_runtime.h>

// ---------------------------------------------------------------------------
// Problem constants (from reference)
// ---------------------------------------------------------------------------
static constexpr int H = 64, W = 64, C = 3, K = 32, PAD = 10;
static constexpr int OH = 53;                 // H + 2*PAD - K + 1
static constexpr int L  = OH * OH;            // 2809 patches
static constexpr int D  = C * K * K;          // 3072
static constexpr int NMEM = 20000;
static constexpr int Lp = 2816;               // pad rows to multiple of 16 (176 tiles)
static constexpr int Np = 20096;              // pad keys to multiple of 128 (1256 tiles)
static constexpr int MT = Lp / 16;            // 176 M tiles
static constexpr int NT = Np / 16;            // 1256 N tiles
static constexpr int KS = D / 32;             // 96 K-steps of 32

typedef __attribute__((ext_vector_type(16))) __bf16 v16bf;
typedef __attribute__((ext_vector_type(8)))  float  v8f;

__device__ inline v8f wmma_bf16(v16bf a, v16bf b, v8f c) {
    return __builtin_amdgcn_wmma_f32_16x16x32_bf16(
        /*neg_a=*/false, a, /*neg_b=*/false, b,
        /*c_mod=*/(short)0, c, /*reuse_a=*/false, /*reuse_b=*/false);
}

__device__ inline unsigned int orderF(float f) {
    unsigned int u = __float_as_uint(f);
    return (u & 0x80000000u) ? ~u : (u | 0x80000000u);   // monotone float->uint
}

__device__ inline unsigned long long shfl_xor_u64(unsigned long long v, int mask) {
    int lo = __shfl_xor((int)(unsigned)(v & 0xffffffffull), mask, 32);
    int hi = __shfl_xor((int)(unsigned)(v >> 32), mask, 32);
    return ((unsigned long long)(unsigned)hi << 32) | (unsigned)(unsigned int)lo;
}

__device__ inline void atomicMaxF(float* addr, float v) {
    if (v >= 0.0f) atomicMax((int*)addr, __float_as_int(v));
    else           atomicMin((unsigned int*)addr, __float_as_uint(v));
}

// ---------------------------------------------------------------------------
// Kernel 0: init best[] (packed +inf) and global max (-inf)
// ---------------------------------------------------------------------------
__global__ void init_state(unsigned long long* best, float* gmax) {
    int i = blockIdx.x * 256 + threadIdx.x;
    if (i < Lp) best[i] = ~0ull;
    if (i == 0) *((unsigned int*)gmax) = 0xFF800000u;     // -inf
}

// ---------------------------------------------------------------------------
// Kernel 1: unfold image -> A fragments (bf16 hi/lo) in WMMA lane layout.
// One thread per (mt, kk, lane); writes 16 bf16 elements per array.
// A 16x32 bf16 layout: lanes 0-15 hold M=lane; e=0..7 -> K=e(+8 if lane>=16),
// e=8..15 -> K=e+8(+8 if lane>=16).  K(e,l) = e + (e&8) + 8*(l>>4).
// ---------------------------------------------------------------------------
__global__ void prep_q(const float* __restrict__ img,
                       __bf16* __restrict__ Ahi, __bf16* __restrict__ Alo) {
    int gid = blockIdx.x * 256 + threadIdx.x;          // (mt*KS + kk)*32 + lane
    if (gid >= MT * KS * 32) return;
    int lane = gid & 31;
    int kk   = (gid >> 5) % KS;
    int mt   = (gid >> 5) / KS;
    int p    = mt * 16 + (lane & 15);                  // patch row
    int oy = p / OH, ox = p % OH;
    v16bf vh, vl;
#pragma unroll
    for (int e = 0; e < 16; ++e) {
        int Kidx = kk * 32 + e + (e & 8) + ((lane >> 4) << 3);
        float x = 0.0f;
        if (p < L) {
            int c = Kidx >> 10, rem = Kidx & 1023, ky = rem >> 5, kx = rem & 31;
            int y = oy + ky - PAD, xx = ox + kx - PAD;
            if ((unsigned)y < (unsigned)H && (unsigned)xx < (unsigned)W)
                x = img[(y * W + xx) * C + c];
        }
        __bf16 hi = (__bf16)x;
        vh[e] = hi;
        vl[e] = (__bf16)(x - (float)hi);
    }
    *(v16bf*)(Ahi + (size_t)gid * 16) = vh;
    *(v16bf*)(Alo + (size_t)gid * 16) = vl;
}

// ---------------------------------------------------------------------------
// Kernel 2: keys -> B fragments (bf16 hi/lo) in WMMA lane layout.
// B 32x16 bf16 layout: same K order as A.
// K(e,l) = e + (e&8) + 8*(l>>4); col = nt*16 + (l&15).
// ---------------------------------------------------------------------------
__global__ void prep_k(const float* __restrict__ keys,
                       __bf16* __restrict__ Bhi, __bf16* __restrict__ Blo) {
    int gid = blockIdx.x * 256 + threadIdx.x;          // (nt*KS + kk)*32 + lane
    if (gid >= NT * KS * 32) return;
    int lane = gid & 31;
    int kk   = (gid >> 5) % KS;
    int nt   = (gid >> 5) / KS;
    int key  = nt * 16 + (lane & 15);
    int kbase = kk * 32 + ((lane >> 4) << 3);
    v16bf vh, vl;
#pragma unroll
    for (int e = 0; e < 16; ++e) {
        float x = (key < NMEM) ? keys[(size_t)key * D + kbase + e + (e & 8)] : 0.0f;
        __bf16 hi = (__bf16)x;
        vh[e] = hi;
        vl[e] = (__bf16)(x - (float)hi);
    }
    *(v16bf*)(Bhi + (size_t)gid * 16) = vh;
    *(v16bf*)(Blo + (size_t)gid * 16) = vl;
}

// ---------------------------------------------------------------------------
// Kernel 3: per-key squared norms (exact fp32, deterministic tree reduction)
// ---------------------------------------------------------------------------
__global__ void key_norms(const float* __restrict__ keys, float* __restrict__ knorm) {
    int j = blockIdx.x;
    __shared__ float red[256];
    float s = 0.0f;
    if (j < NMEM) {
        const float* row = keys + (size_t)j * D;
        for (int i = threadIdx.x; i < D; i += 256) { float v = row[i]; s += v * v; }
    }
    red[threadIdx.x] = s;
    __syncthreads();
    for (int st = 128; st > 0; st >>= 1) {
        if (threadIdx.x < st) red[threadIdx.x] += red[threadIdx.x + st];
        __syncthreads();
    }
    if (threadIdx.x == 0)
        knorm[j] = (j < NMEM) ? red[0] : __int_as_float(0x7f800000);  // +inf pads
}

// ---------------------------------------------------------------------------
// Kernel 4: GEMM (3xBF16 error-compensated) + fused argmin.
// Block = 8 waves (2 M-waves x 4 N-waves); block tile M=64, N=128.
// grid.x = 157 key chunks (keys stay L2-hot), grid.y = 4 M splits (11 iters).
// Each wave: 2x2 accumulator tiles, 12 v_wmma_f32_16x16x32_bf16 per K-step.
// ---------------------------------------------------------------------------
__global__ __launch_bounds__(256)
void gemm_argmin(const __bf16* __restrict__ Ahi, const __bf16* __restrict__ Alo,
                 const __bf16* __restrict__ Bhi, const __bf16* __restrict__ Blo,
                 const float* __restrict__ knorm,
                 unsigned long long* __restrict__ best) {
    const int lane = threadIdx.x & 31;
    const int wv   = threadIdx.x >> 5;
    const int wm   = wv >> 2;            // 0..1
    const int wn   = wv & 3;             // 0..3
    const int ntBase = blockIdx.x * 8;

    float kn[2];
#pragma unroll
    for (int nf = 0; nf < 2; ++nf)
        kn[nf] = knorm[(ntBase + wn * 2 + nf) * 16 + (lane & 15)];

    for (int it = 0; it < 11; ++it) {
        const int mt0 = (blockIdx.y * 11 + it) * 4;
        v8f acc[2][2] = {};

        for (int kk = 0; kk < KS; ++kk) {
            v16bf ah[2], al[2], bh[2], bl[2];
#pragma unroll
            for (int mf = 0; mf < 2; ++mf) {
                size_t off = ((size_t)((mt0 + wm * 2 + mf) * KS + kk) * 32 + lane) * 16;
                ah[mf] = *(const v16bf*)(Ahi + off);
                al[mf] = *(const v16bf*)(Alo + off);
            }
#pragma unroll
            for (int nf = 0; nf < 2; ++nf) {
                size_t off = ((size_t)((ntBase + wn * 2 + nf) * KS + kk) * 32 + lane) * 16;
                bh[nf] = *(const v16bf*)(Bhi + off);
                bl[nf] = *(const v16bf*)(Blo + off);
            }
#pragma unroll
            for (int mf = 0; mf < 2; ++mf)
#pragma unroll
                for (int nf = 0; nf < 2; ++nf) {
                    acc[mf][nf] = wmma_bf16(ah[mf], bh[nf], acc[mf][nf]); // hi*hi
                    acc[mf][nf] = wmma_bf16(ah[mf], bl[nf], acc[mf][nf]); // hi*lo
                    acc[mf][nf] = wmma_bf16(al[mf], bh[nf], acc[mf][nf]); // lo*hi
                }
        }

        // score = ||k||^2 - 2 q.k ; pack (ordered<<32)|idx, min-reduce, atomic.
#pragma unroll
        for (int mf = 0; mf < 2; ++mf) {
            int rowbase = (mt0 + wm * 2 + mf) * 16 + ((lane >> 4) << 3);
#pragma unroll
            for (int r = 0; r < 8; ++r) {
                unsigned long long pk = ~0ull;
#pragma unroll
                for (int nf = 0; nf < 2; ++nf) {
                    float sc = __builtin_fmaf(-2.0f, acc[mf][nf][r], kn[nf]);
                    unsigned col = (unsigned)((ntBase + wn * 2 + nf) * 16 + (lane & 15));
                    unsigned long long p = ((unsigned long long)orderF(sc) << 32) | col;
                    pk = p < pk ? p : pk;
                }
                // min across the 16 lanes of each half (rows differ per half)
                for (int m = 1; m <= 8; m <<= 1) {
                    unsigned long long o = shfl_xor_u64(pk, m);
                    pk = o < pk ? o : pk;
                }
                if ((lane & 15) == 0)
                    atomicMin(&best[rowbase + r], pk);
            }
        }
    }
}

// ---------------------------------------------------------------------------
// Kernel 5: gather values + fold (deterministic, race-free) + global max
// One block per output pixel; sums <=3072 contributions per pixel.
// ---------------------------------------------------------------------------
__global__ void fold_gather(const unsigned long long* __restrict__ best,
                            const float* __restrict__ vals,
                            float* __restrict__ acc, float* __restrict__ gmax) {
    int pix = blockIdx.x;
    int h = pix >> 6, w = pix & 63;
    float s[3] = {0.0f, 0.0f, 0.0f};
    for (int t = threadIdx.x; t < D; t += 256) {
        int c = t >> 10, rem = t & 1023, ky = rem >> 5, kx = rem & 31;
        int oy = h + PAD - ky, ox = w + PAD - kx;
        if ((unsigned)oy < (unsigned)OH && (unsigned)ox < (unsigned)OH) {
            int p = oy * OH + ox;
            unsigned key = (unsigned)(best[p] & 0xffffffffull);
            s[c] += vals[(size_t)key * D + t];
        }
    }
    __shared__ float red[256];
#pragma unroll
    for (int c = 0; c < 3; ++c) {
        red[threadIdx.x] = s[c];
        __syncthreads();
        for (int st = 128; st > 0; st >>= 1) {
            if (threadIdx.x < st) red[threadIdx.x] += red[threadIdx.x + st];
            __syncthreads();
        }
        if (threadIdx.x == 0) {
            float v = red[0];
            acc[pix * 3 + c] = v;
            atomicMaxF(gmax, v);
        }
        __syncthreads();
    }
}

// ---------------------------------------------------------------------------
// Kernel 6: normalize into (H,W,C) output
// ---------------------------------------------------------------------------
__global__ void normalize_out(const float* __restrict__ acc,
                              const float* __restrict__ gmax,
                              float* __restrict__ out) {
    int i = blockIdx.x * 256 + threadIdx.x;
    if (i < H * W * C) out[i] = acc[i] / *gmax;
}

// ---------------------------------------------------------------------------
extern "C" void kernel_launch(void* const* d_in, const int* in_sizes, int n_in,
                              void* d_out, int out_size, void* d_ws, size_t ws_size,
                              hipStream_t stream) {
    const float* img  = (const float*)d_in[0];   // (64,64,3) f32
    const float* keys = (const float*)d_in[1];   // (20000,3072) f32
    const float* vals = (const float*)d_in[2];   // (20000,3072) f32
    float* out = (float*)d_out;                  // (64,64,3) f32

    char* ws = (char*)d_ws;
    size_t off = 0;
    auto carve = [&](size_t bytes) -> void* {
        void* p = ws + off;
        off = (off + bytes + 255) & ~(size_t)255;
        return p;
    };
    __bf16* Ahi = (__bf16*)carve((size_t)MT * KS * 512 * 2);   // 17.3 MB
    __bf16* Alo = (__bf16*)carve((size_t)MT * KS * 512 * 2);   // 17.3 MB
    __bf16* Bhi = (__bf16*)carve((size_t)NT * KS * 512 * 2);   // 123.5 MB
    __bf16* Blo = (__bf16*)carve((size_t)NT * KS * 512 * 2);   // 123.5 MB
    float* knorm = (float*)carve((size_t)Np * 4);
    unsigned long long* best = (unsigned long long*)carve((size_t)Lp * 8);
    float* acc  = (float*)carve((size_t)H * W * C * 4);
    float* gmax = (float*)carve(256);

    init_state<<<(Lp + 255) / 256, 256, 0, stream>>>(best, gmax);
    prep_q<<<(MT * KS * 32) / 256, 256, 0, stream>>>(img, Ahi, Alo);
    prep_k<<<(NT * KS * 32) / 256, 256, 0, stream>>>(keys, Bhi, Blo);
    key_norms<<<Np, 256, 0, stream>>>(keys, knorm);
    gemm_argmin<<<dim3(NT / 8, 4), 256, 0, stream>>>(Ahi, Alo, Bhi, Blo, knorm, best);
    fold_gather<<<H * W, 256, 0, stream>>>(best, vals, acc, gmax);
    normalize_out<<<(H * W * C + 255) / 256, 256, 0, stream>>>(acc, gmax, out);
}
